// FocusedAttnV3_65859028517419
// MI455X (gfx1250) — compile-verified, matrix operand order checked by reading, not run on hardware
//
#include <hip/hip_runtime.h>

// ---------------------------------------------------------------------------
// CDNA5 (gfx1250, wave32) fused attention:
//   kp = k@Wk.T ; vp = v@Wv.T ; qp = (q@Wk.T)*scale   (reference uses Wk for q!)
//   attn = softmax(qp.kp^T + mask) ; x = attn.vp ; out = x@Wproj.T + bproj
// All GEMMs via v_wmma_f32_16x16x32_bf16. Flash-style attention, no NxN matrix.
// Pass 0 pre-converts activations+weights to bf16 so GEMM inner loops are pure
// global_load_b128 + v_wmma (no per-wave redundant cvt).
// ---------------------------------------------------------------------------

typedef __attribute__((ext_vector_type(16))) __bf16       v16bf;
typedef __attribute__((ext_vector_type(8)))  float        v8f;
typedef __attribute__((ext_vector_type(4)))  int          i4v;
typedef __attribute__((ext_vector_type(4)))  float        f4v;
typedef __attribute__((ext_vector_type(4)))  unsigned int u4v;

#define DIM_C   1024
#define HEADS_C 16
#define HD_C    64
#define NQ_C    1024
#define NKV_C   4096
#define B_C     2
#define SCALE_C 0.125f          // HEAD_DIM^-0.5 = 1/8
#define NEG_BIG (-1.0e30f)      // stands in for -inf (keeps exp() NaN-free)

union FragB16 {
  v16bf  v;
  __bf16 e[16];
  u4v    u[2];
};

// A/B 16-bit fragment layout (ISA 7.12.2, 16-bit A 16x32):
//   lane: row/col = lane&15 ; kbase = 8*(lane>>4)
//   elements 0..7  -> K = kbase .. kbase+7
//   elements 8..15 -> K = 16+kbase .. 16+kbase+7
// (dense B assumed mirror of A: lane = N column, same K striping)
static __device__ __forceinline__ v16bf frag_from_bf16(const __bf16* base, int kb) {
  FragB16 f;
  f.u[0] = *(const u4v*)(base + kb);        // 8 bf16, 16B
  f.u[1] = *(const u4v*)(base + kb + 16);   // 8 bf16, 16B
  return f.v;
}

static __device__ __forceinline__ v8f wmma_bf16(v16bf a, v16bf b, v8f c) {
  return __builtin_amdgcn_wmma_f32_16x16x32_bf16(false, a, false, b,
                                                 (short)0, c, false, false);
}

// ---------------------------------------------------------------------------
// Streaming f32 -> bf16 convert (8 elems/thread). Sources are dead after this
// pass -> non-temporal loads; bf16 results stay cache-resident for the GEMMs.
// ---------------------------------------------------------------------------
__global__ void __launch_bounds__(256)
cvt_f32_bf16(const float* __restrict__ src, __bf16* __restrict__ dst, int n8)
{
  const int i = (int)(blockIdx.x * blockDim.x + threadIdx.x);
  if (i >= n8) return;
  const float* p = src + (size_t)i * 8;
  f4v a = __builtin_nontemporal_load((const f4v*)p);
  f4v b = __builtin_nontemporal_load((const f4v*)(p + 4));
  union { __bf16 e[8]; u4v u; } pk;
#pragma unroll
  for (int j = 0; j < 4; ++j) {
    pk.e[j]     = (__bf16)a[j];
    pk.e[4 + j] = (__bf16)b[j];
  }
  *(u4v*)(dst + (size_t)i * 8) = pk.u;
}

// ---------------------------------------------------------------------------
// Projection GEMM (all-bf16 operands):
//   Y[m][n] = scale * sum_k X[m][k] * W[n][k]
// One wave -> 16x64 output tile (4 WMMA accumulators, A fragment reused 4x).
// transposed==0: Y head-major  [b][h][pos][64]   (qp, kp)
// transposed==1: Y d-major     [b][h][d][Nrows]  (vp; packed 16B stores)
// ---------------------------------------------------------------------------
__global__ void __launch_bounds__(256)
proj_gemm_bf16(const __bf16* __restrict__ X, const __bf16* __restrict__ W,
               __bf16* __restrict__ Y, int Mrows, int Nrows,
               float scale, int transposed)
{
  const int lane = threadIdx.x & 31;
  const int wave = (int)((blockIdx.x * blockDim.x + threadIdx.x) >> 5);
  const int ntg  = DIM_C / 64;          // 16 column groups
  const int mt   = wave / ntg;
  const int nt   = wave - mt * ntg;
  const int m_base = mt * 16;
  if (m_base >= Mrows) return;
  const int n_base = nt * 64;
  const int rc  = lane & 15;
  const int hlf = lane >> 4;
  const int kb  = hlf << 3;

  v8f acc[4] = {};
  const __bf16* xrow = X + (size_t)(m_base + rc) * DIM_C;
  const __bf16* wrow[4];
#pragma unroll
  for (int t = 0; t < 4; ++t)
    wrow[t] = W + (size_t)(n_base + t * 16 + rc) * DIM_C;

  for (int k0 = 0; k0 < DIM_C; k0 += 32) {
    v16bf a = frag_from_bf16(xrow + k0, kb);
#pragma unroll
    for (int t = 0; t < 4; ++t)
      acc[t] = wmma_bf16(a, frag_from_bf16(wrow[t] + k0, kb), acc[t]);
  }

  // C/D layout: VGPR r, lane -> (M = r + 8*hlf, N = lane&15)
#pragma unroll
  for (int t = 0; t < 4; ++t) {
    const int n = n_base + t * 16 + rc;
    const int h = n >> 6;
    const int d = n & 63;
    if (!transposed) {
#pragma unroll
      for (int r = 0; r < 8; ++r) {
        const int m   = m_base + (hlf << 3) + r;
        const int bb  = m / Nrows;
        const int pos = m - bb * Nrows;
        const size_t idx = (((size_t)(bb * HEADS_C + h) * Nrows + pos) << 6) + d;
        Y[idx] = (__bf16)(acc[t][r] * scale);
      }
    } else {
      union { __bf16 e[8]; u4v u; } pk;
#pragma unroll
      for (int r = 0; r < 8; ++r) pk.e[r] = (__bf16)(acc[t][r] * scale);
      const int m   = m_base + (hlf << 3);
      const int bb  = m / Nrows;
      const int pos = m - bb * Nrows;      // 8 consecutive positions -> one 16B store
      const size_t idx = ((size_t)(bb * HEADS_C + h) * HD_C + d) * (size_t)Nrows + pos;
      *(u4v*)(Y + idx) = pk.u;
    }
  }
}

// ---------------------------------------------------------------------------
// Flash attention. One wave per (b, h, 16-row q tile); 128 iterations of kv32.
// Trick: compute S' = Kp_tile(A) x Qp^T(B) so the C layout (lane = q column,
// kv striped over VGPRs) is *exactly* the A-fragment layout needed to feed
// P = exp(S'-m) into the P.V WMMA — no LDS transpose.
// Masks streamed once with non-temporal loads (they are the 512MB of traffic).
// ---------------------------------------------------------------------------
__global__ void __launch_bounds__(256)
flash_attn_bf16(const __bf16* __restrict__ qp,   // [b][h][NQ][64]
                const __bf16* __restrict__ kp,   // [b][h][NKV][64]
                const __bf16* __restrict__ vpT,  // [b][h][64][NKV]
                const int*    __restrict__ am,   // [h][NQ][NKV]
                const float*  __restrict__ dm,   // [h][NQ][NKV]
                __bf16* __restrict__ xo)         // [b][NQ][DIM]
{
  const int lane = threadIdx.x & 31;
  const int wave = (int)((blockIdx.x * blockDim.x + threadIdx.x) >> 5);
  const int qtiles = NQ_C / 16;                   // 64
  const int bb  = wave / (HEADS_C * qtiles);
  const int rem = wave - bb * (HEADS_C * qtiles);
  const int h   = rem / qtiles;
  const int qt  = rem - h * qtiles;
  const int q_base = qt * 16;
  const int qc  = lane & 15;                      // this lane's q column
  const int hlf = lane >> 4;
  const int kb  = hlf << 3;

  const __bf16* qb   = qp  + (((size_t)(bb * HEADS_C + h) * NQ_C) << 6);
  const __bf16* kbse = kp  + (((size_t)(bb * HEADS_C + h) * NKV_C) << 6);
  const __bf16* vbse = vpT + ((size_t)(bb * HEADS_C + h) * HD_C) * NKV_C;
  const int*    arow = am  + ((size_t)h * NQ_C + q_base + qc) * NKV_C;
  const float*  drow = dm  + ((size_t)h * NQ_C + q_base + qc) * NKV_C;

  // Qp as B operand (lane = q column), loaded once: two d-chunks of K=32
  const __bf16* qrow = qb + ((size_t)(q_base + qc) << 6);
  const v16bf bq0 = frag_from_bf16(qrow,      kb);   // d 0..31
  const v16bf bq1 = frag_from_bf16(qrow + 32, kb);   // d 32..63

  float m_run = NEG_BIG, l_run = 0.0f;
  v8f acc[4] = {};                                   // O tile, 4 x (16q x 16d)

  for (int kv0 = 0; kv0 < NKV_C; kv0 += 32) {
    __builtin_prefetch(arow + kv0 + 32, 0, 0);       // global_prefetch
    __builtin_prefetch(drow + kv0 + 32, 0, 0);

    // S' tiles: lane holds column q = qc, rows kv = 8*hlf + r (+16 for s1)
    const __bf16* krow0 = kbse + ((size_t)(kv0 + qc) << 6);
    const __bf16* krow1 = krow0 + (16 << 6);
    v8f s0 = {}, s1 = {};
    s0 = wmma_bf16(frag_from_bf16(krow0,      kb), bq0, s0);
    s0 = wmma_bf16(frag_from_bf16(krow0 + 32, kb), bq1, s0);
    s1 = wmma_bf16(frag_from_bf16(krow1,      kb), bq0, s1);
    s1 = wmma_bf16(frag_from_bf16(krow1 + 32, kb), bq1, s1);

    // masks for (q = qc, kv = kv0 + 8*hlf + r [+16]) — streamed, non-temporal
    int im[16]; float fm[16];
    const int*   ap = arow + kv0 + (hlf << 3);
    const float* dp = drow + kv0 + (hlf << 3);
    *(i4v*)(im + 0)  = __builtin_nontemporal_load((const i4v*)(ap));
    *(i4v*)(im + 4)  = __builtin_nontemporal_load((const i4v*)(ap + 4));
    *(i4v*)(im + 8)  = __builtin_nontemporal_load((const i4v*)(ap + 16));
    *(i4v*)(im + 12) = __builtin_nontemporal_load((const i4v*)(ap + 20));
    *(f4v*)(fm + 0)  = __builtin_nontemporal_load((const f4v*)(dp));
    *(f4v*)(fm + 4)  = __builtin_nontemporal_load((const f4v*)(dp + 4));
    *(f4v*)(fm + 8)  = __builtin_nontemporal_load((const f4v*)(dp + 16));
    *(f4v*)(fm + 12) = __builtin_nontemporal_load((const f4v*)(dp + 20));

    float sv0[8], sv1[8];
#pragma unroll
    for (int r = 0; r < 8; ++r) {
      sv0[r] = s0[r] + ((im[r]     > 0) ? NEG_BIG : 0.0f) + fm[r];
      sv1[r] = s1[r] + ((im[8 + r] > 0) ? NEG_BIG : 0.0f) + fm[8 + r];
    }

    // online softmax: all 16 lane-values belong to q = qc; combine halves xor16
    float cmax = NEG_BIG;
#pragma unroll
    for (int r = 0; r < 8; ++r) cmax = fmaxf(cmax, fmaxf(sv0[r], sv1[r]));
    cmax = fmaxf(cmax, __shfl_xor(cmax, 16, 32));
    const float m_new = fmaxf(m_run, cmax);
    const float corr  = __expf(m_run - m_new);

    FragB16 pf;                                      // P as A operand directly
    float csum = 0.0f;
#pragma unroll
    for (int r = 0; r < 8; ++r) {
      const float p0 = __expf(sv0[r] - m_new);
      const float p1 = __expf(sv1[r] - m_new);
      csum += p0 + p1;
      pf.e[r]     = (__bf16)p0;                      // K = kb + r
      pf.e[8 + r] = (__bf16)p1;                      // K = 16 + kb + r
    }
    csum += __shfl_xor(csum, 16, 32);
    l_run = l_run * corr + csum;
    m_run = m_new;

    // rescale O rows (acc row r is q = 8*hlf + r -> fetch that q's corr)
    float fr[8];
#pragma unroll
    for (int r = 0; r < 8; ++r) fr[r] = __shfl(corr, (hlf << 3) + r, 32);
#pragma unroll
    for (int t = 0; t < 4; ++t)
#pragma unroll
      for (int r = 0; r < 8; ++r) acc[t][r] *= fr[r];

    // O += P.V : B operand from d-major Vp (lane = d column, kv contiguous)
#pragma unroll
    for (int t = 0; t < 4; ++t) {
      const __bf16* vrow = vbse + (size_t)(t * 16 + qc) * NKV_C + kv0;
      acc[t] = wmma_bf16(pf.v, frag_from_bf16(vrow, kb), acc[t]);
    }
  }

  // normalize + store x[b][q][h*64+d]
  const float inv = 1.0f / l_run;
  float ir[8];
#pragma unroll
  for (int r = 0; r < 8; ++r) ir[r] = __shfl(inv, (hlf << 3) + r, 32);
#pragma unroll
  for (int t = 0; t < 4; ++t)
#pragma unroll
    for (int r = 0; r < 8; ++r) {
      const int qr = q_base + (hlf << 3) + r;
      const size_t idx = ((size_t)bb * NQ_C + qr) * DIM_C + h * HD_C + t * 16 + qc;
      xo[idx] = (__bf16)(acc[t][r] * ir[r]);
    }
}

// ---------------------------------------------------------------------------
// Output projection: out[m][n] = sum_k x[m][k]*Wproj[n][k] + bproj[n]  (f32 out)
// ---------------------------------------------------------------------------
__global__ void __launch_bounds__(256)
out_proj_gemm(const __bf16* __restrict__ Xa, const __bf16* __restrict__ W,
              const float* __restrict__ bias, float* __restrict__ Y)
{
  const int lane = threadIdx.x & 31;
  const int wave = (int)((blockIdx.x * blockDim.x + threadIdx.x) >> 5);
  const int ntg  = DIM_C / 64;
  const int mt   = wave / ntg;
  const int nt   = wave - mt * ntg;
  const int m_base = mt * 16;
  if (m_base >= B_C * NQ_C) return;
  const int n_base = nt * 64;
  const int rc  = lane & 15;
  const int hlf = lane >> 4;
  const int kb  = hlf << 3;

  v8f acc[4] = {};
  const __bf16* xrow = Xa + (size_t)(m_base + rc) * DIM_C;
  const __bf16* wrow[4];
#pragma unroll
  for (int t = 0; t < 4; ++t)
    wrow[t] = W + (size_t)(n_base + t * 16 + rc) * DIM_C;

  for (int k0 = 0; k0 < DIM_C; k0 += 32) {
    v16bf a = frag_from_bf16(xrow + k0, kb);
#pragma unroll
    for (int t = 0; t < 4; ++t)
      acc[t] = wmma_bf16(a, frag_from_bf16(wrow[t] + k0, kb), acc[t]);
  }
#pragma unroll
  for (int t = 0; t < 4; ++t) {
    const int n  = n_base + t * 16 + rc;
    const float bv = bias[n];
#pragma unroll
    for (int r = 0; r < 8; ++r) {
      const int m = m_base + (hlf << 3) + r;
      Y[(size_t)m * DIM_C + n] = acc[t][r] + bv;
    }
  }
}

// ---------------------------------------------------------------------------
extern "C" void kernel_launch(void* const* d_in, const int* in_sizes, int n_in,
                              void* d_out, int out_size, void* d_ws, size_t ws_size,
                              hipStream_t stream) {
  (void)in_sizes; (void)n_in; (void)out_size; (void)ws_size;
  const float* q  = (const float*)d_in[0];
  const float* k  = (const float*)d_in[1];
  const float* v  = (const float*)d_in[2];
  const int*   am = (const int*)  d_in[3];
  const float* dm = (const float*)d_in[4];
  const float* Wk = (const float*)d_in[5];
  const float* Wv = (const float*)d_in[6];
  const float* Wp = (const float*)d_in[7];
  const float* bp = (const float*)d_in[8];
  float* out = (float*)d_out;

  // workspace layout (bf16 elems):
  //  qbf 2M | kbf 8M | vbf 8M | wkbf 1M | wvbf 1M | wpbf 1M |
  //  qp 2M | kp 8M | vpT 8M | xa 2M                      = 41M elems = 82 MB
  const size_t NQD  = (size_t)B_C * NQ_C  * DIM_C;   // 2M
  const size_t NKVD = (size_t)B_C * NKV_C * DIM_C;   // 8M
  const size_t WD   = (size_t)DIM_C * DIM_C;         // 1M
  __bf16* qbf  = (__bf16*)d_ws;
  __bf16* kbf  = qbf  + NQD;
  __bf16* vbf  = kbf  + NKVD;
  __bf16* wkbf = vbf  + NKVD;
  __bf16* wvbf = wkbf + WD;
  __bf16* wpbf = wvbf + WD;
  __bf16* qp   = wpbf + WD;
  __bf16* kp   = qp   + NQD;
  __bf16* vpT  = kp   + NKVD;
  __bf16* xa   = vpT  + NKVD;

  const int thr = 256, wpb = thr / 32;

  // pass 0: one-time f32 -> bf16 conversion (kills redundant per-wave cvt)
  cvt_f32_bf16<<<(int)(NQD  / 8 / thr), thr, 0, stream>>>(q,  qbf,  (int)(NQD  / 8));
  cvt_f32_bf16<<<(int)(NKVD / 8 / thr), thr, 0, stream>>>(k,  kbf,  (int)(NKVD / 8));
  cvt_f32_bf16<<<(int)(NKVD / 8 / thr), thr, 0, stream>>>(v,  vbf,  (int)(NKVD / 8));
  cvt_f32_bf16<<<(int)(WD   / 8 / thr), thr, 0, stream>>>(Wk, wkbf, (int)(WD   / 8));
  cvt_f32_bf16<<<(int)(WD   / 8 / thr), thr, 0, stream>>>(Wv, wvbf, (int)(WD   / 8));
  cvt_f32_bf16<<<(int)(WD   / 8 / thr), thr, 0, stream>>>(Wp, wpbf, (int)(WD   / 8));

  const int waves_q = (B_C * NQ_C  / 16) * (DIM_C / 64);   // 2048
  const int waves_k = (B_C * NKV_C / 16) * (DIM_C / 64);   // 8192
  const int waves_a = B_C * HEADS_C * (NQ_C / 16);         // 2048
  const int waves_o = (B_C * NQ_C  / 16) * (DIM_C / 64);   // 2048

  // NOTE: reference projects q with Wk (q_proj unused) — faithful.
  proj_gemm_bf16<<<waves_q / wpb, thr, 0, stream>>>(qbf, wkbf, qp,  B_C * NQ_C,  NQ_C,  SCALE_C, 0);
  proj_gemm_bf16<<<waves_k / wpb, thr, 0, stream>>>(kbf, wkbf, kp,  B_C * NKV_C, NKV_C, 1.0f,    0);
  proj_gemm_bf16<<<waves_k / wpb, thr, 0, stream>>>(vbf, wvbf, vpT, B_C * NKV_C, NKV_C, 1.0f,    1);
  flash_attn_bf16<<<waves_a / wpb, thr, 0, stream>>>(qp, kp, vpT, am, dm, xa);
  out_proj_gemm<<<waves_o / wpb, thr, 0, stream>>>(xa, wpbf, bp, out);
}
